// PointNet_71683004171100
// MI455X (gfx1250) — compile-verified, matrix operand order checked by reading, not run on hardware
//
#include <hip/hip_runtime.h>
#include <math.h>
#include <stdint.h>

// ---------------------------------------------------------------------------
// Types for CDNA5 WMMA (wave32): v_wmma_f32_16x16x32_bf16
// ---------------------------------------------------------------------------
typedef __attribute__((ext_vector_type(16))) __bf16 v16bf;
typedef __attribute__((ext_vector_type(8)))  __bf16 v8bf;
typedef __attribute__((ext_vector_type(8)))  float  v8f;

#define ASTR 40    // padded LDS row stride (elements) for A tile (64 x 32 bf16)
#define BSTR 40    // padded LDS row stride (elements) for B tile (128 x 32 bf16)
#define CSTR 136   // padded LDS row stride for C tile (64 x 128 bf16)
#define HGK  288   // padded K (bf16 row stride) of hgin [S, 257 -> 288]

__device__ __forceinline__ float lrelu(float v) { return v > 0.f ? v : 0.01f * v; }

// monotone mapping float -> uint so atomicMax(uint) == float max
__device__ __forceinline__ unsigned fmono(float f) {
  unsigned b = __float_as_uint(f);
  return (b & 0x80000000u) ? ~b : (b | 0x80000000u);
}
__device__ __forceinline__ float funmono(unsigned u) {
  return (u & 0x80000000u) ? __uint_as_float(u & 0x7fffffffu) : __uint_as_float(~u);
}

// ---------------------------------------------------------------------------
// Async global->LDS 16B copy (CDNA5 GLOBAL_LOAD_ASYNC_TO_LDS_B128).
// Builtin signature (from toolchain diagnostics): param1 = AS(1) int4*,
// param2 = AS(3) int4*, then imm offset, imm cpol.
// ---------------------------------------------------------------------------
#if __has_builtin(__builtin_amdgcn_global_load_async_to_lds_b128)
#define HAVE_ASYNC_LDS 1
typedef int vi4 __attribute__((vector_size(16)));
typedef __attribute__((address_space(1))) vi4* as1_vi4p;
typedef __attribute__((address_space(3))) vi4* as3_vi4p;
__device__ __forceinline__ void async_cp16(const void* g, void* l) {
  // generic LDS pointers carry the LDS offset in the low 32 bits
  __builtin_amdgcn_global_load_async_to_lds_b128(
      (as1_vi4p)(uintptr_t)g, (as3_vi4p)(unsigned)(uintptr_t)l, 0, 0);
}
__device__ __forceinline__ void async_wait0() {
#if __has_builtin(__builtin_amdgcn_s_wait_asynccnt)
  __builtin_amdgcn_s_wait_asynccnt(0);
#else
  asm volatile("s_wait_asynccnt 0x0" ::: "memory");
#endif
}
#endif

// ---------------------------------------------------------------------------
// Segment geometry kernels (quad-processing: idx is sorted, so 4 consecutive
// points usually share a segment -> 1 atomic instead of 4)
// ---------------------------------------------------------------------------
__global__ void seg_accum(const float* __restrict__ pos, const int* __restrict__ idx,
                          float* csum, float* ccnt, int N) {
  int i = (blockIdx.x * 256 + threadIdx.x) * 4;
  if (i + 3 < N) {
    int4 s4 = *(const int4*)(idx + i);
    float4 p0 = *(const float4*)(pos + (size_t)i * 3);
    float4 p1 = *(const float4*)(pos + (size_t)i * 3 + 4);
    float4 p2 = *(const float4*)(pos + (size_t)i * 3 + 8);
    float X[4] = {p0.x, p0.w, p1.z, p2.y};
    float Y[4] = {p0.y, p1.x, p1.w, p2.z};
    float Z[4] = {p0.z, p1.y, p2.x, p2.w};
    int   s[4] = {s4.x, s4.y, s4.z, s4.w};
    if (s[0] == s[3]) {
      atomicAdd(&csum[s[0] * 3 + 0], X[0] + X[1] + X[2] + X[3]);
      atomicAdd(&csum[s[0] * 3 + 1], Y[0] + Y[1] + Y[2] + Y[3]);
      atomicAdd(&csum[s[0] * 3 + 2], Z[0] + Z[1] + Z[2] + Z[3]);
      atomicAdd(&ccnt[s[0]], 4.0f);
    } else {
#pragma unroll
      for (int j = 0; j < 4; ++j) {
        atomicAdd(&csum[s[j] * 3 + 0], X[j]);
        atomicAdd(&csum[s[j] * 3 + 1], Y[j]);
        atomicAdd(&csum[s[j] * 3 + 2], Z[j]);
        atomicAdd(&ccnt[s[j]], 1.0f);
      }
    }
  } else if (i < N) {
    for (int j = i; j < N; ++j) {
      int s = idx[j];
      atomicAdd(&csum[s * 3 + 0], pos[(size_t)j * 3 + 0]);
      atomicAdd(&csum[s * 3 + 1], pos[(size_t)j * 3 + 1]);
      atomicAdd(&csum[s * 3 + 2], pos[(size_t)j * 3 + 2]);
      atomicAdd(&ccnt[s], 1.0f);
    }
  }
}

__global__ void seg_center(float* csum, const float* ccnt, int S) {
  int s = blockIdx.x * 256 + threadIdx.x;
  if (s < S) {
    float inv = 1.0f / fmaxf(ccnt[s], 1.0f);
    csum[s * 3 + 0] *= inv;
    csum[s * 3 + 1] *= inv;
    csum[s * 3 + 2] *= inv;
  }
}

__global__ void seg_r2(const float* __restrict__ pos, const int* __restrict__ idx,
                       const float* __restrict__ center, float* r2, int N) {
  int i = (blockIdx.x * 256 + threadIdx.x) * 4;
  if (i + 3 < N) {
    int4 s4 = *(const int4*)(idx + i);
    float4 p0 = *(const float4*)(pos + (size_t)i * 3);
    float4 p1 = *(const float4*)(pos + (size_t)i * 3 + 4);
    float4 p2 = *(const float4*)(pos + (size_t)i * 3 + 8);
    float X[4] = {p0.x, p0.w, p1.z, p2.y};
    float Y[4] = {p0.y, p1.x, p1.w, p2.z};
    float Z[4] = {p0.z, p1.y, p2.x, p2.w};
    int   s[4] = {s4.x, s4.y, s4.z, s4.w};
    if (s[0] == s[3]) {
      float c0 = center[s[0] * 3 + 0], c1 = center[s[0] * 3 + 1], c2 = center[s[0] * 3 + 2];
      float m = 0.f;
#pragma unroll
      for (int j = 0; j < 4; ++j) {
        float dx = X[j] - c0, dy = Y[j] - c1, dz = Z[j] - c2;
        m = fmaxf(m, dx * dx + dy * dy + dz * dz);
      }
      atomicMax((int*)&r2[s[0]], __float_as_int(m));
    } else {
#pragma unroll
      for (int j = 0; j < 4; ++j) {
        float dx = X[j] - center[s[j] * 3 + 0];
        float dy = Y[j] - center[s[j] * 3 + 1];
        float dz = Z[j] - center[s[j] * 3 + 2];
        atomicMax((int*)&r2[s[j]], __float_as_int(dx * dx + dy * dy + dz * dz));
      }
    }
  } else if (i < N) {
    for (int j = i; j < N; ++j) {
      int s = idx[j];
      float dx = pos[(size_t)j * 3 + 0] - center[s * 3 + 0];
      float dy = pos[(size_t)j * 3 + 1] - center[s * 3 + 1];
      float dz = pos[(size_t)j * 3 + 2] - center[s * 3 + 2];
      atomicMax((int*)&r2[s], __float_as_int(dx * dx + dy * dy + dz * dz));
    }
  }
}

__global__ void seg_fin(const float* r2, const float* ccnt, float* rinv, float* diam, int S) {
  int s = blockIdx.x * 256 + threadIdx.x;
  if (s < S) {
    float r  = sqrtf(fmaxf(r2[s], 0.f));
    float rs = (r > 0.f) ? r : 1.0f;
    rinv[s] = 1.0f / rs;
    diam[s] = (ccnt[s] > 0.f) ? 2.0f * rs : 0.f;
  }
}

__global__ void pool_init(unsigned* pool, int n) {
  int i = blockIdx.x * 256 + threadIdx.x;
  if (i < n) pool[i] = 0x00800000u;  // fmono(-FLT_MAX)
}

// pooled features + diameter -> bf16, zero-padded to K=288 (pure-copy GEMM input)
__global__ void build_hgin(const unsigned* __restrict__ pool, const float* __restrict__ ccnt,
                           const float* __restrict__ diam, __bf16* hgin, int S) {
  int i = blockIdx.x * 256 + threadIdx.x;
  if (i < S * HGK) {
    int s = i / HGK, c = i - s * HGK;
    float v = 0.f;
    if (c < 256)        v = (ccnt[s] > 0.f) ? funmono(pool[(size_t)s * 256 + c]) : 0.f;
    else if (c == 256)  v = diam[s];
    hgin[i] = (__bf16)v;
  }
}

// one-time weight prep: W[k][n] f32 -> Wt[n][kpad] bf16, zero pad k>=kact
__global__ void wt_prep(const float* __restrict__ W, __bf16* Wt, int ncols, int kact, int kpad) {
  int i = blockIdx.x * 256 + threadIdx.x;
  if (i < ncols * kpad) {
    int n = i / kpad, k = i - n * kpad;
    Wt[i] = (k < kact) ? (__bf16)W[(size_t)k * ncols + n] : (__bf16)0.f;
  }
}

// fold BN stats (64 partial slots) into scale/shift
__global__ void bn_fin(const float* __restrict__ gs, const float* __restrict__ gq,
                       const float* __restrict__ g, const float* __restrict__ be,
                       float* scale, float* shift, int ncols, float invN) {
  int c = blockIdx.x * 128 + threadIdx.x;
  if (c < ncols) {
    float s = 0.f, q = 0.f;
    for (int j = 0; j < 64; ++j) { s += gs[j * ncols + c]; q += gq[j * ncols + c]; }
    float mu  = s * invN;
    float var = fmaxf(q * invN - mu * mu, 0.f);
    float inv = rsqrtf(var + 1e-5f);
    float sc  = g[c] * inv;
    scale[c] = sc;
    shift[c] = be[c] - mu * sc;
  }
}

__global__ void final_out(const float* __restrict__ yg2, const float* __restrict__ sc,
                          const float* __restrict__ sh, float* out, int n) {
  int i = blockIdx.x * 256 + threadIdx.x;
  if (i < n) {
    int c = i & 511;
    out[i] = lrelu(fmaf(sc[c], yg2[i], sh[c]));
  }
}

// ---------------------------------------------------------------------------
// Fused WMMA GEMM.  MODE: 0=L1 (concat(x,pos_n)@W1 -> y1 bf16 + stats)
//                         1=L2 stats pass (BN1+lrelu(y1))@W2 -> stats only
//                         2=L2 pool pass  -> BN2+lrelu -> segmented max pool
//                         3=G1 (hgin@GW1 -> yg1 + stats)
//                         4=G2 ((BN(yg1)+lrelu)@GW2 -> yg2 + stats)
// Block: 256 threads = 8 waves, computes 64 rows x 128 cols; waves 4(M) x 2(N),
// each wave: 16x64 output = 4 x v_wmma_f32_16x16x32_bf16 per K-chunk of 32.
// B-side weights are pre-transposed bf16 [n][KCH*32]; staging is a pure copy,
// done with GLOBAL_LOAD_ASYNC_TO_LDS_B128 when available.
// ---------------------------------------------------------------------------
struct GP {
  const float* x; const float* pos; const float* center; const float* rinv;
  const int* idx;
  const __bf16* y1r; __bf16* y1w;
  const float* ain; const __bf16* hginb;
  const __bf16* Wt; const float* bias;
  const float* scaleIn; const float* shiftIn;
  const float* scaleOut; const float* shiftOut;
  float* gsum; float* gsq;
  float* yout;
  unsigned* pool;
  int rows; int ncols;
};

template <int MODE, int KCH>
__global__ __launch_bounds__(256) void gemm_k(GP P) {
  __shared__ __bf16 Ash[64 * ASTR];
  __shared__ __bf16 Bsh[128 * BSTR];
  __shared__ float csum_sh[128], csq_sh[128];
  __shared__ float scsh[512], shsh[512];  // BN params for A-side (modes 1/2/4)

  const int tid = threadIdx.x;
  const int lane = tid & 31, wave = tid >> 5;
  const int wy = wave >> 1, wx = wave & 1;
  const int lane15 = lane & 15, hh = lane >> 4;
  const int row0 = blockIdx.x * 64;
  const int nbase = blockIdx.y * 128;
  const int KPAD = KCH * 32;

  if (tid < 128) { csum_sh[tid] = 0.f; csq_sh[tid] = 0.f; }
  if constexpr (MODE == 1 || MODE == 2 || MODE == 4) {
    for (int i = tid; i < KPAD; i += 256) { scsh[i] = P.scaleIn[i]; shsh[i] = P.shiftIn[i]; }
  }

  // staging coordinates
  const int ar = tid >> 2;            // A: row 0..63
  const int akg = (tid & 3) * 8;      // A: k-group (8 wide)

  v8f acc[4] = {};

  for (int kc = 0; kc < KCH; ++kc) {
    __syncthreads();
    const int k0 = kc * 32;

    // ---- stage B: pure bf16 copy from pre-transposed weights Wt[n][KPAD]
#pragma unroll
    for (int i = 0; i < 2; ++i) {
      int e = i * 256 + tid;
      int n = e >> 2, kg = (e & 3) * 8;
      const __bf16* gsrc = P.Wt + (size_t)(nbase + n) * KPAD + k0 + kg;
      __bf16* ldst = Bsh + n * BSTR + kg;
#ifdef HAVE_ASYNC_LDS
      async_cp16(gsrc, ldst);
#else
      *(v8bf*)ldst = *(const v8bf*)gsrc;
#endif
    }

    // ---- stage A: one 8-wide k group per thread
    {
      const int row = row0 + ar;
      const int k = k0 + akg;
      const bool rowok = (row < P.rows);

      if constexpr (MODE == 3) {  // pure bf16 copy (padded, zero-filled)
        if (rowok) {
#ifdef HAVE_ASYNC_LDS
          async_cp16(P.hginb + (size_t)row * HGK + k, Ash + ar * ASTR + akg);
#else
          *(v8bf*)(Ash + ar * ASTR + akg) = *(const v8bf*)(P.hginb + (size_t)row * HGK + k);
#endif
        } else {
          v8bf o;
#pragma unroll
          for (int j = 0; j < 8; ++j) o[j] = (__bf16)0.f;
          *(v8bf*)(Ash + ar * ASTR + akg) = o;
        }
      } else {
        float f[8];
#pragma unroll
        for (int j = 0; j < 8; ++j) f[j] = 0.f;

        if constexpr (MODE == 0) {
          if (rowok && (k + 8 <= 64)) {
            float4 a0 = *(const float4*)(P.x + (size_t)row * 64 + k);
            float4 a1 = *(const float4*)(P.x + (size_t)row * 64 + k + 4);
            f[0] = a0.x; f[1] = a0.y; f[2] = a0.z; f[3] = a0.w;
            f[4] = a1.x; f[5] = a1.y; f[6] = a1.z; f[7] = a1.w;
          } else if (rowok) {
#pragma unroll
            for (int j = 0; j < 8; ++j) {
              int kk = k + j;
              if (kk < 64) f[j] = P.x[(size_t)row * 64 + kk];
              else if (kk < 67) {
                int s = P.idx[row]; int d = kk - 64;
                f[j] = (P.pos[(size_t)row * 3 + d] - P.center[s * 3 + d]) * P.rinv[s];
              }
            }
          }
        } else if constexpr (MODE == 1 || MODE == 2) {
          if (rowok) {
            v8bf y = *(const v8bf*)(P.y1r + (size_t)row * 128 + k);
#pragma unroll
            for (int j = 0; j < 8; ++j) f[j] = lrelu(fmaf(scsh[k + j], (float)y[j], shsh[k + j]));
            if (kc + 1 < KCH) __builtin_prefetch(P.y1r + (size_t)row * 128 + k + 32, 0, 1);
          }
        } else {  // MODE 4
          if (rowok) {
            float4 a0 = *(const float4*)(P.ain + (size_t)row * 512 + k);
            float4 a1 = *(const float4*)(P.ain + (size_t)row * 512 + k + 4);
            float g[8] = {a0.x, a0.y, a0.z, a0.w, a1.x, a1.y, a1.z, a1.w};
#pragma unroll
            for (int j = 0; j < 8; ++j) f[j] = lrelu(fmaf(scsh[k + j], g[j], shsh[k + j]));
            if (kc + 1 < KCH) __builtin_prefetch(P.ain + (size_t)row * 512 + k + 32, 0, 1);
          }
        }
        v8bf o;
#pragma unroll
        for (int j = 0; j < 8; ++j) o[j] = (__bf16)f[j];
        *(v8bf*)(Ash + ar * ASTR + akg) = o;
      }
    }

#ifdef HAVE_ASYNC_LDS
    async_wait0();  // drain this wave's async LDS writes before the barrier
#endif
    __syncthreads();

    // ---- fragments + WMMA
    // A (16x32 bf16): lane m=lane15, half hh; K runs [hh*8,+8) and [16+hh*8,+8)
    const __bf16* ap = &Ash[(wy * 16 + lane15) * ASTR + hh * 8];
    v8bf alo = *(const v8bf*)ap;
    v8bf ahi = *(const v8bf*)(ap + 16);
    v16bf a;
#pragma unroll
    for (int i = 0; i < 8; ++i) { a[i] = alo[i]; a[i + 8] = ahi[i]; }

#pragma unroll
    for (int t = 0; t < 4; ++t) {
      const __bf16* bp = &Bsh[(wx * 64 + t * 16 + lane15) * BSTR + hh * 8];
      v8bf blo = *(const v8bf*)bp;
      v8bf bhi = *(const v8bf*)(bp + 16);
      v16bf b;
#pragma unroll
      for (int i = 0; i < 8; ++i) { b[i] = blo[i]; b[i + 8] = bhi[i]; }
      acc[t] = __builtin_amdgcn_wmma_f32_16x16x32_bf16(
          false, a, false, b, (short)0, acc[t], false, false);
    }
  }

  // ---- consumers -----------------------------------------------------------
  if constexpr (MODE == 0) {
    // C tile -> LDS, then coalesced bf16 row stores; stats from registers
    __shared__ __bf16 Csh[64 * CSTR];
#pragma unroll
    for (int t = 0; t < 4; ++t) {
      int cl = wx * 64 + t * 16 + lane15;
      float s8 = 0.f, q8 = 0.f;
#pragma unroll
      for (int v = 0; v < 8; ++v) {
        int m = wy * 16 + hh * 8 + v;          // C layout: M = v + 8*half
        float val = acc[t][v] + P.bias[cl];
        Csh[m * CSTR + cl] = (__bf16)val;
        if (row0 + m < P.rows) { s8 += val; q8 += val * val; }
      }
      atomicAdd(&csum_sh[cl], s8);
      atomicAdd(&csq_sh[cl], q8);
    }
    __syncthreads();
    {
      int r = tid >> 2, part = tid & 3;
      int row = row0 + r;
      if (row < P.rows) {
#pragma unroll
        for (int j = 0; j < 4; ++j)
          *(v8bf*)(P.y1w + (size_t)row * 128 + part * 32 + j * 8) =
              *(const v8bf*)(Csh + r * CSTR + part * 32 + j * 8);
      }
    }
    int slot = blockIdx.x & 63;
    if (tid < 128) atomicAdd(&P.gsum[(size_t)slot * 128 + tid], csum_sh[tid]);
    else           atomicAdd(&P.gsq[(size_t)slot * 128 + (tid - 128)], csq_sh[tid - 128]);
  }

  if constexpr (MODE == 1 || MODE == 3 || MODE == 4) {
#pragma unroll
    for (int t = 0; t < 4; ++t) {
      int cl = wx * 64 + t * 16 + lane15;
      int cg = nbase + cl;
      float s8 = 0.f, q8 = 0.f;
#pragma unroll
      for (int v = 0; v < 8; ++v) {
        int row = row0 + wy * 16 + hh * 8 + v;
        if (row < P.rows) {
          float val = acc[t][v] + P.bias[cg];
          if constexpr (MODE == 3 || MODE == 4) P.yout[(size_t)row * P.ncols + cg] = val;
          s8 += val; q8 += val * val;
        }
      }
      atomicAdd(&csum_sh[cl], s8);
      atomicAdd(&csq_sh[cl], q8);
    }
    __syncthreads();
    int slot = blockIdx.x & 63;
    if (tid < 128) atomicAdd(&P.gsum[(size_t)slot * P.ncols + nbase + tid], csum_sh[tid]);
    else           atomicAdd(&P.gsq[(size_t)slot * P.ncols + nbase + (tid - 128)], csq_sh[tid - 128]);
  }

  if constexpr (MODE == 2) {
    __shared__ float h2sh[64 * 129];
    __shared__ int idxsh[64];
    if (tid < 64) idxsh[tid] = (row0 + tid < P.rows) ? P.idx[row0 + tid] : -1;
#pragma unroll
    for (int t = 0; t < 4; ++t) {
      int cl = wx * 64 + t * 16 + lane15;
      int cg = nbase + cl;
#pragma unroll
      for (int v = 0; v < 8; ++v) {
        int m = wy * 16 + hh * 8 + v;
        float val = acc[t][v] + P.bias[cg];
        val = lrelu(fmaf(P.scaleOut[cg], val, P.shiftOut[cg]));
        h2sh[m * 129 + cl] = val;
      }
    }
    __syncthreads();
    if (tid < 128) {
      int c = tid;
      int cur = -1; float cm = 0.f;
      for (int r = 0; r < 64; ++r) {
        if (row0 + r >= P.rows) break;
        int s = idxsh[r];
        float v = h2sh[r * 129 + c];
        if (s != cur) {
          if (cur >= 0) atomicMax(&P.pool[(size_t)cur * 256 + nbase + c], fmono(cm));
          cur = s; cm = v;
        } else cm = fmaxf(cm, v);
      }
      if (cur >= 0) atomicMax(&P.pool[(size_t)cur * 256 + nbase + c], fmono(cm));
    }
  }
}

// ---------------------------------------------------------------------------
// Host launcher
// ---------------------------------------------------------------------------
extern "C" void kernel_launch(void* const* d_in, const int* in_sizes, int n_in,
                              void* d_out, int out_size, void* d_ws, size_t ws_size,
                              hipStream_t stream) {
  const float* pos = (const float*)d_in[0];
  const float* x   = (const float*)d_in[1];
  const int*   idx = (const int*)d_in[2];
  const float* W1  = (const float*)d_in[4];
  const float* b1  = (const float*)d_in[5];
  const float* g1  = (const float*)d_in[6];
  const float* be1 = (const float*)d_in[7];
  const float* W2  = (const float*)d_in[8];
  const float* b2  = (const float*)d_in[9];
  const float* g2  = (const float*)d_in[10];
  const float* be2 = (const float*)d_in[11];
  const float* GW1  = (const float*)d_in[12];
  const float* Gb1  = (const float*)d_in[13];
  const float* Gg1  = (const float*)d_in[14];
  const float* Gbe1 = (const float*)d_in[15];
  const float* GW2  = (const float*)d_in[16];
  const float* Gb2  = (const float*)d_in[17];
  const float* Gg2  = (const float*)d_in[18];
  const float* Gbe2 = (const float*)d_in[19];

  const int N = in_sizes[0] / 3;
  const int S = out_size / 512;

  // workspace layout (256B aligned chunks)
  size_t off = 0;
  auto alloc = [&](size_t bytes) { size_t r = off; off = (off + bytes + 255) & ~(size_t)255; return r; };
  char* ws = (char*)d_ws;
  auto F = [&](size_t o) { return (float*)(ws + o); };
  auto H = [&](size_t o) { return (__bf16*)(ws + o); };

  size_t o_csum = alloc((size_t)S * 3 * 4);
  size_t o_ccnt = alloc((size_t)S * 4);
  size_t o_r2   = alloc((size_t)S * 4);
  size_t o_sum1 = alloc(64 * 128 * 4), o_sq1 = alloc(64 * 128 * 4);
  size_t o_sum2 = alloc(64 * 256 * 4), o_sq2 = alloc(64 * 256 * 4);
  size_t o_sg1  = alloc(64 * 512 * 4), o_qg1 = alloc(64 * 512 * 4);
  size_t o_sg2  = alloc(64 * 512 * 4), o_qg2 = alloc(64 * 512 * 4);
  size_t zbytes = off;                               // zero-init region
  size_t o_rinv = alloc((size_t)S * 4);
  size_t o_diam = alloc((size_t)S * 4);
  size_t o_sc1 = alloc(128 * 4), o_sh1 = alloc(128 * 4);
  size_t o_sc2 = alloc(256 * 4), o_sh2 = alloc(256 * 4);
  size_t o_scG1 = alloc(512 * 4), o_shG1 = alloc(512 * 4);
  size_t o_scG2 = alloc(512 * 4), o_shG2 = alloc(512 * 4);
  size_t o_w1t  = alloc(128 * 96 * 2);               // bf16 transposed weights
  size_t o_w2t  = alloc(256 * 128 * 2);
  size_t o_gw1t = alloc(512 * 288 * 2);
  size_t o_gw2t = alloc(512 * 512 * 2);
  size_t o_hgin = alloc((size_t)S * HGK * 2);        // bf16
  size_t o_pool = alloc((size_t)S * 256 * 4);
  size_t o_yg1  = alloc((size_t)S * 512 * 4);
  size_t o_yg2  = alloc((size_t)S * 512 * 4);
  size_t o_y1   = alloc((size_t)N * 128 * 2);        // bf16

  (void)hipMemsetAsync(d_ws, 0, zbytes, stream);
  pool_init<<<(S * 256 + 255) / 256, 256, 0, stream>>>((unsigned*)(ws + o_pool), S * 256);

  // one-time weight transpose/convert
  wt_prep<<<(128 * 96 + 255) / 256, 256, 0, stream>>>(W1, H(o_w1t), 128, 67, 96);
  wt_prep<<<(256 * 128 + 255) / 256, 256, 0, stream>>>(W2, H(o_w2t), 256, 128, 128);
  wt_prep<<<(512 * 288 + 255) / 256, 256, 0, stream>>>(GW1, H(o_gw1t), 512, 257, 288);
  wt_prep<<<(512 * 512 + 255) / 256, 256, 0, stream>>>(GW2, H(o_gw2t), 512, 512, 512);

  // segment geometry
  const int qb = ((N + 3) / 4 + 255) / 256;
  seg_accum<<<qb, 256, 0, stream>>>(pos, idx, F(o_csum), F(o_ccnt), N);
  seg_center<<<(S + 255) / 256, 256, 0, stream>>>(F(o_csum), F(o_ccnt), S);
  seg_r2<<<qb, 256, 0, stream>>>(pos, idx, F(o_csum), F(o_r2), N);
  seg_fin<<<(S + 255) / 256, 256, 0, stream>>>(F(o_r2), F(o_ccnt), F(o_rinv), F(o_diam), S);

  const int mb = (N + 63) / 64;   // row blocks for local MLP
  const int sb = (S + 63) / 64;   // row blocks for global MLP

  // local layer 1: [N,67]@[67,128] -> y1 (bf16) + BN1 stats
  GP p{}; p.x = x; p.pos = pos; p.center = F(o_csum); p.rinv = F(o_rinv); p.idx = idx;
  p.y1w = H(o_y1);
  p.Wt = H(o_w1t); p.bias = b1; p.gsum = F(o_sum1); p.gsq = F(o_sq1);
  p.rows = N; p.ncols = 128;
  gemm_k<0, 3><<<dim3(mb, 1), 256, 0, stream>>>(p);
  bn_fin<<<1, 128, 0, stream>>>(F(o_sum1), F(o_sq1), g1, be1, F(o_sc1), F(o_sh1), 128, 1.0f / N);

  // local layer 2, pass A (stats): (BN1+lrelu(y1))@W2 -> BN2 stats
  GP q{}; q.y1r = H(o_y1); q.scaleIn = F(o_sc1); q.shiftIn = F(o_sh1);
  q.Wt = H(o_w2t); q.bias = b2; q.gsum = F(o_sum2); q.gsq = F(o_sq2);
  q.rows = N; q.ncols = 256;
  gemm_k<1, 4><<<dim3(mb, 2), 256, 0, stream>>>(q);
  bn_fin<<<2, 128, 0, stream>>>(F(o_sum2), F(o_sq2), g2, be2, F(o_sc2), F(o_sh2), 256, 1.0f / N);

  // local layer 2, pass B: recompute, BN2+lrelu, segmented max pool
  GP r{}; r.y1r = H(o_y1); r.scaleIn = F(o_sc1); r.shiftIn = F(o_sh1);
  r.Wt = H(o_w2t); r.bias = b2; r.scaleOut = F(o_sc2); r.shiftOut = F(o_sh2);
  r.idx = idx; r.pool = (unsigned*)(ws + o_pool);
  r.rows = N; r.ncols = 256;
  gemm_k<2, 4><<<dim3(mb, 2), 256, 0, stream>>>(r);

  build_hgin<<<(S * HGK + 255) / 256, 256, 0, stream>>>(
      (const unsigned*)(ws + o_pool), F(o_ccnt), F(o_diam), H(o_hgin), S);

  // global layer 1: [S,257]@[257,512] -> yg1 + stats
  GP u{}; u.hginb = H(o_hgin); u.Wt = H(o_gw1t); u.bias = Gb1;
  u.gsum = F(o_sg1); u.gsq = F(o_qg1); u.yout = F(o_yg1);
  u.rows = S; u.ncols = 512;
  gemm_k<3, 9><<<dim3(sb, 4), 256, 0, stream>>>(u);
  bn_fin<<<4, 128, 0, stream>>>(F(o_sg1), F(o_qg1), Gg1, Gbe1, F(o_scG1), F(o_shG1), 512, 1.0f / S);

  // global layer 2: (BN+lrelu(yg1))@[512,512] -> yg2 + stats
  GP v{}; v.ain = F(o_yg1); v.scaleIn = F(o_scG1); v.shiftIn = F(o_shG1);
  v.Wt = H(o_gw2t); v.bias = Gb2; v.gsum = F(o_sg2); v.gsq = F(o_qg2); v.yout = F(o_yg2);
  v.rows = S; v.ncols = 512;
  gemm_k<4, 16><<<dim3(sb, 4), 256, 0, stream>>>(v);
  bn_fin<<<4, 128, 0, stream>>>(F(o_sg2), F(o_qg2), Gg2, Gbe2, F(o_scG2), F(o_shG2), 512, 1.0f / S);

  final_out<<<(S * 512 + 255) / 256, 256, 0, stream>>>(
      F(o_yg2), F(o_scG2), F(o_shG2), (float*)d_out, S * 512);
}
